// TopKGateParallel_33990371180785
// MI455X (gfx1250) — compile-verified, matrix operand order checked by reading, not run on hardware
//
#include <hip/hip_runtime.h>
#include <hip/hip_bf16.h>

typedef __attribute__((ext_vector_type(16))) __bf16 v16bf;
typedef __attribute__((ext_vector_type(8)))  __bf16 v8bf;
typedef __attribute__((ext_vector_type(8)))  float  v8f;

#define B_   8
#define S_   4096
#define D_   2048
#define E_   64
#define NTOK (B_ * S_)

// Pack 8 consecutive f32 (32B aligned) into bf16 lanes [BASE, BASE+8) of a v16bf.
template <int BASE>
__device__ inline void pack8(v16bf& dst, const float* __restrict__ p) {
    const float4* q = reinterpret_cast<const float4*>(p);
    float4 a = q[0];
    float4 b = q[1];
    dst[BASE + 0] = (__bf16)a.x; dst[BASE + 1] = (__bf16)a.y;
    dst[BASE + 2] = (__bf16)a.z; dst[BASE + 3] = (__bf16)a.w;
    dst[BASE + 4] = (__bf16)b.x; dst[BASE + 5] = (__bf16)b.y;
    dst[BASE + 6] = (__bf16)b.z; dst[BASE + 7] = (__bf16)b.w;
}

// Load a B fragment (16 bf16 per lane = two 16B runs) straight from bf16 memory.
__device__ inline v16bf load_bfrag(const __bf16* __restrict__ row, int r0, int r1) {
    v8bf lo = *reinterpret_cast<const v8bf*>(row + r0);
    v8bf hi = *reinterpret_cast<const v8bf*>(row + r1);
    return __builtin_shufflevector(lo, hi, 0, 1, 2, 3, 4, 5, 6, 7,
                                           8, 9, 10, 11, 12, 13, 14, 15);
}

// ---------------------------------------------------------------------------
// Kernel 0: one-time gate_w f32 [64,2048] -> bf16 [64,2048] (256 KB, L2-hot).
// ---------------------------------------------------------------------------
__global__ __launch_bounds__(256) void gate_w_cvt_kernel(
    const float* __restrict__ gate_w, __bf16* __restrict__ gate_w_bf) {
    const int i = (blockIdx.x * blockDim.x + threadIdx.x) * 4;
    float4 v = *reinterpret_cast<const float4*>(gate_w + i);
    v8bf o;  // write 8B per thread: 4 bf16
    __bf16 b0 = (__bf16)v.x, b1 = (__bf16)v.y, b2 = (__bf16)v.z, b3 = (__bf16)v.w;
    __bf16* p = gate_w_bf + i;
    p[0] = b0; p[1] = b1; p[2] = b2; p[3] = b3;
    (void)o;
}

// ---------------------------------------------------------------------------
// Kernel 1: logits[NTOK, 64] = x[NTOK, 2048] @ gate_w[64, 2048]^T  (bf16 WMMA)
// Each wave32 owns a 16-token x 64-expert tile: 4 v8f accumulators,
// K marched in steps of 32 with v_wmma_f32_16x16x32_bf16.
// A (x) is converted f32->bf16 on the fly; B comes pre-converted in bf16.
// ---------------------------------------------------------------------------
__global__ __launch_bounds__(256) void gate_gemm_kernel(
    const float*  __restrict__ x,
    const __bf16* __restrict__ gate_w_bf,
    float* __restrict__ logits) {

    const int wave = threadIdx.x >> 5;
    const int lane = threadIdx.x & 31;
    const int half = lane >> 4;     // which 16-lane half of the wave
    const int mn   = lane & 15;     // row (A) / column (B) index within tile
    const int tok0 = blockIdx.x * 128 + wave * 16;

    v8f acc0 = {}, acc1 = {}, acc2 = {}, acc3 = {};

    const float*  __restrict__ arow  = x         + (size_t)(tok0 + mn) * D_;
    const __bf16* __restrict__ brow0 = gate_w_bf + (size_t)( 0 + mn) * D_;
    const __bf16* __restrict__ brow1 = gate_w_bf + (size_t)(16 + mn) * D_;
    const __bf16* __restrict__ brow2 = gate_w_bf + (size_t)(32 + mn) * D_;
    const __bf16* __restrict__ brow3 = gate_w_bf + (size_t)(48 + mn) * D_;

    for (int k = 0; k < D_; k += 32) {
        const int r0 = k + half * 8;        // first contiguous 8-elem run
        const int r1 = k + 16 + half * 8;   // second contiguous 8-elem run

        // prefetch the x stream two K-chunks ahead (global_prefetch_b8)
        if (k + 64 < D_) __builtin_prefetch(arow + k + 64, 0, 1);

        v16bf a;
        pack8<0>(a, arow + r0);
        pack8<8>(a, arow + r1);

        v16bf b0 = load_bfrag(brow0, r0, r1);
        v16bf b1 = load_bfrag(brow1, r0, r1);
        v16bf b2 = load_bfrag(brow2, r0, r1);
        v16bf b3 = load_bfrag(brow3, r0, r1);

        acc0 = __builtin_amdgcn_wmma_f32_16x16x32_bf16(false, a, false, b0, (short)0, acc0, false, false);
        acc1 = __builtin_amdgcn_wmma_f32_16x16x32_bf16(false, a, false, b1, (short)0, acc1, false, false);
        acc2 = __builtin_amdgcn_wmma_f32_16x16x32_bf16(false, a, false, b2, (short)0, acc2, false, false);
        acc3 = __builtin_amdgcn_wmma_f32_16x16x32_bf16(false, a, false, b3, (short)0, acc3, false, false);
    }

    // C/D layout: VGPR j, lanes 0-15 -> token j, lanes 16-31 -> token j+8; N = mn.
    #pragma unroll
    for (int j = 0; j < 8; ++j) {
        float* row = logits + (size_t)(tok0 + half * 8 + j) * E_;
        row[ 0 + mn] = acc0[j];
        row[16 + mn] = acc1[j];
        row[32 + mn] = acc2[j];
        row[48 + mn] = acc3[j];
    }
}

// ---------------------------------------------------------------------------
// Kernel 2: one thread per token. Clean softmax -> LB-loss partial sums
// (LDS f32 atomics, then 64 global f32 atomics per block), noisy top-2,
// top-2 softmax scattered into a zeroed 64-wide row, ids out.
// ---------------------------------------------------------------------------
__global__ __launch_bounds__(256) void gate_topk_kernel(
    const float* __restrict__ logits,
    const float* __restrict__ noise_weight,
    const float* __restrict__ noise,
    float* __restrict__ out_w,
    int*   __restrict__ out_ids,
    float* __restrict__ accum) {

    __shared__ float sacc[E_];
    const int tid = blockIdx.x * blockDim.x + threadIdx.x;
    if (threadIdx.x < E_) sacc[threadIdx.x] = 0.0f;
    __syncthreads();

    float l[E_];
    const float* __restrict__ lrow = logits + (size_t)tid * E_;
    #pragma unroll
    for (int e = 0; e < E_; e += 4) {
        float4 v = *reinterpret_cast<const float4*>(lrow + e);
        l[e] = v.x; l[e + 1] = v.y; l[e + 2] = v.z; l[e + 3] = v.w;
    }

    // Clean softmax over all 64 experts (for load-balance loss mean).
    float m = l[0];
    #pragma unroll
    for (int e = 1; e < E_; ++e) m = fmaxf(m, l[e]);
    float s = 0.0f;
    #pragma unroll
    for (int e = 0; e < E_; ++e) s += __expf(l[e] - m);
    const float inv = 1.0f / s;
    #pragma unroll
    for (int e = 0; e < E_; ++e) atomicAdd(&sacc[e], __expf(l[e] - m) * inv);

    // Noisy logits -> top-2.
    const float* __restrict__ nrow = noise + (size_t)tid * E_;
    float v0 = -__builtin_inff(), v1 = -__builtin_inff();
    int   i0 = 0, i1 = 0;
    #pragma unroll
    for (int e = 0; e < E_; ++e) {
        float nv = l[e] + nrow[e] * noise_weight[e];
        if (nv > v0) { v1 = v0; i1 = i0; v0 = nv; i0 = e; }
        else if (nv > v1) { v1 = nv; i1 = e; }
    }

    // softmax over the two selected values, scattered into zeros.
    const float w0 = 1.0f / (1.0f + __expf(v1 - v0));
    const float w1 = 1.0f - w0;
    float* __restrict__ orow = out_w + (size_t)tid * E_;
    #pragma unroll
    for (int e = 0; e < E_; e += 4) {
        float4 v;
        v.x = (e + 0 == i0) ? w0 : ((e + 0 == i1) ? w1 : 0.0f);
        v.y = (e + 1 == i0) ? w0 : ((e + 1 == i1) ? w1 : 0.0f);
        v.z = (e + 2 == i0) ? w0 : ((e + 2 == i1) ? w1 : 0.0f);
        v.w = (e + 3 == i0) ? w0 : ((e + 3 == i1) ? w1 : 0.0f);
        *reinterpret_cast<float4*>(orow + e) = v;
    }
    out_ids[tid * 2 + 0] = i0;
    out_ids[tid * 2 + 1] = i1;

    __syncthreads();
    if (threadIdx.x < E_) atomicAdd(&accum[threadIdx.x], sacc[threadIdx.x]);
}

// ---------------------------------------------------------------------------
// Kernel 3: loss = mean((accum/NTOK - 1/E)^2) * 0.01
// ---------------------------------------------------------------------------
__global__ void gate_loss_kernel(const float* __restrict__ accum,
                                 float* __restrict__ loss_out) {
    __shared__ float red[E_];
    const int e = threadIdx.x;
    float mean = accum[e] * (1.0f / (float)NTOK);
    float d = mean - (1.0f / (float)E_);
    red[e] = d * d;
    __syncthreads();
    if (e == 0) {
        float t = 0.0f;
        #pragma unroll
        for (int i = 0; i < E_; ++i) t += red[i];
        *loss_out = (t / (float)E_) * 0.01f;
    }
}

extern "C" void kernel_launch(void* const* d_in, const int* in_sizes, int n_in,
                              void* d_out, int out_size, void* d_ws, size_t ws_size,
                              hipStream_t stream) {
    const float* x            = (const float*)d_in[0];
    const float* gate_w       = (const float*)d_in[1];
    const float* noise_weight = (const float*)d_in[2];
    const float* noise        = (const float*)d_in[3];

    float* out      = (float*)d_out;
    float* out_w    = out;                                  // [NTOK, 64]
    int*   out_ids  = (int*)(out + (size_t)NTOK * E_);      // [NTOK, 2]
    float* out_loss = out + (size_t)NTOK * E_ + (size_t)NTOK * 2;  // scalar

    float*  logits    = (float*)d_ws;                       // 8 MB
    float*  accum     = logits + (size_t)NTOK * E_;         // 64 floats
    __bf16* gate_w_bf = (__bf16*)(accum + E_);              // 256 KB

    hipMemsetAsync(accum, 0, E_ * sizeof(float), stream);

    gate_w_cvt_kernel<<<(E_ * D_ / 4) / 256, 256, 0, stream>>>(gate_w, gate_w_bf);
    gate_gemm_kernel<<<NTOK / 128, 256, 0, stream>>>(x, gate_w_bf, logits);
    gate_topk_kernel<<<NTOK / 256, 256, 0, stream>>>(logits, noise_weight, noise,
                                                     out_w, out_ids, accum);
    gate_loss_kernel<<<1, E_, 0, stream>>>(accum, out_loss);
}
